// MessHierEncoder_74766790689056
// MI455X (gfx1250) — matrix operand order repaired and checked by reading, not compiled, not run on hardware
//
#include <hip/hip_runtime.h>
#include <hip/hip_bf16.h>

// Problem constants (match reference)
#define HDIM  256
#define H3DIM 768
#define NT_ 20000
#define MT_ 40000
#define NA_ 40000
#define MA_ 80000

#define USE_ASYNC_LDS 1   // gfx1250 GLOBAL_LOAD_ASYNC_TO_LDS_B128 staging path

typedef __attribute__((ext_vector_type(16))) __bf16 bf16x16;
typedef __attribute__((ext_vector_type(8)))  __bf16 bf16x8;
typedef __attribute__((ext_vector_type(8)))  float  f32x8;

__device__ __forceinline__ float sig_(float x) { return 1.0f / (1.0f + __expf(-x)); }

__device__ __forceinline__ bf16x16 cat16_(bf16x8 lo, bf16x8 hi) {
    return __builtin_shufflevector(lo, hi, 0,1,2,3,4,5,6,7,8,9,10,11,12,13,14,15);
}

#if USE_ASYNC_LDS
// One 16B async copy: memory -> LDS (ASYNCcnt-tracked, no VGPR data round-trip).
__device__ __forceinline__ void async_ld_b128(unsigned lds_off, const void* g) {
    asm volatile("global_load_async_to_lds_b128 %0, %1, off"
                 :: "v"(lds_off), "v"(g) : "memory");
}
#endif

// ---------------------------------------------------------------------------
// Generic bf16 WMMA GEMM:  C[M,N] = epilogue( Aop[M,K] @ W[K,N] + extras )
//   Aop row m = (gidx ? Agat[gidx[m*gs+go]] : A[m])          (bf16, contiguous K)
//   Wt is the PRE-TRANSPOSED weight, [N,K] bf16 (k-contiguous rows)
//   extras: + addm[m, :]  + rowW[rbase + ridx[m*rs+ro], :]  + bias[:]
//   epilogue: optional relu, optional zero of logical row 0; writes f32 and/or bf16
// Block tile 64x64, K-step 64 (2 WMMA k-chunks per LDS tile), 256 threads = 8 waves,
// wave = 16x32 output (two f32 16x16 accumulators, 4 WMMAs per tile).
// Staging: double-buffered global_load_async_to_lds_b128, 4 x 16B per thread per tile;
// steady-state loop prefetches next tile then waits asynccnt<=4 (async loads complete
// in order), final tile peeled with asynccnt<=0. K must be a multiple of 64, >= 128.
// ---------------------------------------------------------------------------
#define BM 64
#define BN 64
#define BK 64
#define SAD 72             // LDS A row stride (halves), keeps 16B alignment
#define SWD 72             // LDS W^T row stride (halves)
#define BUFH (BM*SAD + BN*SWD)

__global__ __launch_bounds__(256) void k_gemm(
    const __bf16* __restrict__ A,
    const __bf16* __restrict__ Agat,
    const int*    __restrict__ gidx, int g_stride, int g_off,
    const __bf16* __restrict__ Wt,           // [N,K] pre-transposed
    const float*  __restrict__ addm, int ld_add,
    const float*  __restrict__ rowW, int ld_rowW,
    const int*    __restrict__ ridx, int r_stride, int r_off, int r_base,
    const float*  __restrict__ bias,
    const __bf16* __restrict__ zpage,        // >=32B of zeros for OOB rows
    float*        __restrict__ Cf,
    __bf16*       __restrict__ Cb, int ldc,
    int M, int N, int K, int relu, int mask0)
{
    __shared__ __bf16 smem[2 * BUFH];

    const int bm   = blockIdx.x * BM;
    const int bn   = blockIdx.y * BN;
    const int t    = threadIdx.x;
    const int lane = t & 31;
    const int wave = t >> 5;
    const int msub = wave & 3;   // 0..3 -> M sub-tile of 16
    const int nsub = wave >> 2;  // 0..1 -> N sub-tile of 32

    f32x8 acc0 = {};
    f32x8 acc1 = {};

    // tile loader mapping: thread -> (row 0..63, 2 x 8-half k-chunks); same for A and Wt
    const int lrow  = t >> 2;
    const int lcol2 = (t & 3) * 16;          // first of two adjacent 8-half chunks
    const int gm_l  = bm + lrow;
    const bool aok  = (gm_l < M);
    const __bf16* abase = zpage;
    if (aok) {
        if (gidx) abase = Agat + (size_t)gidx[(size_t)gm_l * g_stride + g_off] * K;
        else      abase = A    + (size_t)gm_l * K;
    }
    const __bf16* wrow = Wt + (size_t)(bn + lrow) * K;

    // fragment loaders (per LDS buffer base)
    auto lda = [&](const __bf16* sA, int c) -> bf16x16 {
        const int base = (msub * 16 + (lane & 15)) * SAD + c * 32 + (lane >> 4) * 8;
        return cat16_(*(const bf16x8*)(&sA[base]), *(const bf16x8*)(&sA[base + 16]));
    };
    auto ldb = [&](const __bf16* sW, int j, int c) -> bf16x16 {
        const int base = (nsub * 32 + j * 16 + (lane & 15)) * SWD + c * 32 + (lane >> 4) * 8;
        return cat16_(*(const bf16x8*)(&sW[base]), *(const bf16x8*)(&sW[base + 16]));
    };
    auto compute = [&](const __bf16* buf) {
        const __bf16* sA = buf;
        const __bf16* sW = buf + BM * SAD;
        bf16x16 a0 = lda(sA, 0);
        bf16x16 a1 = lda(sA, 1);
        bf16x16 b00 = ldb(sW, 0, 0);
        bf16x16 b10 = ldb(sW, 1, 0);
        acc0 = __builtin_amdgcn_wmma_f32_16x16x32_bf16(false, a0, false, b00, (short)0, acc0, false, false);
        acc1 = __builtin_amdgcn_wmma_f32_16x16x32_bf16(false, a0, false, b10, (short)0, acc1, false, false);
        bf16x16 b01 = ldb(sW, 0, 1);
        bf16x16 b11 = ldb(sW, 1, 1);
        acc0 = __builtin_amdgcn_wmma_f32_16x16x32_bf16(false, a1, false, b01, (short)0, acc0, false, false);
        acc1 = __builtin_amdgcn_wmma_f32_16x16x32_bf16(false, a1, false, b11, (short)0, acc1, false, false);
    };

#if USE_ASYNC_LDS
    const unsigned lds0  = (unsigned)(size_t)(&smem[0]);
    const unsigned a_st0 = lds0 + (unsigned)((lrow * SAD + lcol2) * 2);
    const unsigned w_st0 = lds0 + (unsigned)((BM * SAD + lrow * SWD + lcol2) * 2);
    auto issue = [&](int b, int k0) {
        const unsigned boff = (unsigned)(b * (BUFH * 2));
        const void* ga0 = aok ? (const void*)(abase + k0 + lcol2)     : (const void*)zpage;
        const void* ga1 = aok ? (const void*)(abase + k0 + lcol2 + 8) : (const void*)zpage;
        async_ld_b128(a_st0 + boff,      ga0);
        async_ld_b128(a_st0 + boff + 16, ga1);
        async_ld_b128(w_st0 + boff,      (const void*)(wrow + k0 + lcol2));
        async_ld_b128(w_st0 + boff + 16, (const void*)(wrow + k0 + lcol2 + 8));
    };

    issue(0, 0);
    int b = 0, k0 = 0;
    for (; k0 + BK < K; k0 += BK, b ^= 1) {            // steady state: unconditional prefetch
        issue(b ^ 1, k0 + BK);
        asm volatile("s_wait_asynccnt 0x4" ::: "memory");  // current tile's 4 copies done
        __syncthreads();
        compute(smem + b * BUFH);
        __syncthreads();
    }
    asm volatile("s_wait_asynccnt 0x0" ::: "memory");      // peeled final tile
    __syncthreads();
    compute(smem + b * BUFH);
#else
    for (int k0 = 0; k0 < K; k0 += BK) {
        bf16x8 a0 = aok ? *(const bf16x8*)(abase + k0 + lcol2)     : *(const bf16x8*)zpage;
        bf16x8 a1 = aok ? *(const bf16x8*)(abase + k0 + lcol2 + 8) : *(const bf16x8*)zpage;
        *(bf16x8*)(&smem[lrow * SAD + lcol2])     = a0;
        *(bf16x8*)(&smem[lrow * SAD + lcol2 + 8]) = a1;
        bf16x8 w0 = *(const bf16x8*)(wrow + k0 + lcol2);
        bf16x8 w1 = *(const bf16x8*)(wrow + k0 + lcol2 + 8);
        *(bf16x8*)(&smem[BM * SAD + lrow * SWD + lcol2])     = w0;
        *(bf16x8*)(&smem[BM * SAD + lrow * SWD + lcol2 + 8]) = w1;
        __syncthreads();
        compute(smem);
        __syncthreads();
    }
#endif

    // Epilogue. C layout: VGPR i -> M = i (lanes 0-15) or 8+i (lanes 16-31); N = lane%16.
    const int rsel = (lane >> 4) * 8;
    const int col0 = bn + nsub * 32 + (lane & 15);
#pragma unroll
    for (int j = 0; j < 2; ++j) {
        const int gn = col0 + j * 16;
#pragma unroll
        for (int q = 0; q < 8; ++q) {
            const int gm = bm + msub * 16 + rsel + q;
            if (gm >= M) continue;
            float v = (j == 0) ? acc0[q] : acc1[q];
            if (bias) v += bias[gn];
            if (addm) v += addm[(size_t)gm * ld_add + gn];
            if (rowW) v += rowW[(size_t)(r_base + ridx[(size_t)gm * r_stride + r_off]) * ld_rowW + gn];
            if (relu) v = fmaxf(v, 0.0f);
            if (mask0 && gm == 0) v = 0.0f;
            if (Cf) Cf[(size_t)gm * ldc + gn] = v;
            if (Cb) Cb[(size_t)gm * ldc + gn] = (__bf16)v;
        }
    }
}

// ---------------------------------------------------------------------------
// Elementwise / gather kernels
// ---------------------------------------------------------------------------
__global__ void k_zero_f(float* p, long n) {
    long i = (long)blockIdx.x * blockDim.x + threadIdx.x;
    if (i < n) p[i] = 0.0f;
}
__global__ void k_zero_h(__bf16* p, long n) {
    long i = (long)blockIdx.x * blockDim.x + threadIdx.x;
    if (i < n) p[i] = (__bf16)0.0f;
}
// transpose-convert: dst[n*K + k] = (bf16) src[k*N + n]   (src is K rows x N cols, f32)
__global__ void k_cvt_t(const float* __restrict__ src, __bf16* __restrict__ dst, int K, int N) {
    long i = (long)blockIdx.x * blockDim.x + threadIdx.x;
    if (i >= (long)K * N) return;
    int n = (int)(i / K), k = (int)(i % K);
    dst[i] = (__bf16)src[(size_t)k * N + n];
}
// out[m, 0:256] = (bf16) emb[idx[m*stride+off], 0:256]
__global__ void k_gather_rows(const float* __restrict__ emb, const int* __restrict__ idx,
                              int stride, int off, __bf16* __restrict__ out, int rows) {
    long i = (long)blockIdx.x * blockDim.x + threadIdx.x;
    if (i >= (long)rows * HDIM) return;
    int m = (int)(i >> 8), n = (int)(i & 255);
    out[i] = (__bf16)emb[(size_t)idx[m * stride + off] * HDIM + n];
}
// out[m, :] = bf16( sum_d src[idx[m*D+d], :] )   (f32 accumulation)
__global__ void k_gather_sum(const __bf16* __restrict__ src, const int* __restrict__ idx,
                             int D, __bf16* __restrict__ out, int rows) {
    long i = (long)blockIdx.x * blockDim.x + threadIdx.x;
    if (i >= (long)rows * HDIM) return;
    int m = (int)(i >> 8), n = (int)(i & 255);
    float a = 0.0f;
    for (int d = 0; d < D; ++d) a += (float)src[(size_t)idx[m * D + d] * HDIM + n];
    out[i] = (__bf16)a;
}
// Graph-encoder pre-activation: one-hot concat == 3 weight-row gathers + bias
// fmess layout [MA,4]: (src, dst, bond, pos); rows: fnode[src], 64+bond, 68+pos
__global__ void k_zpre_onehot(const float* __restrict__ Wg, const float* __restrict__ bg,
                              const int* __restrict__ fnode, const int* __restrict__ fmess,
                              int ld, float* __restrict__ out, int Mrows) {
    long i = (long)blockIdx.x * blockDim.x + threadIdx.x;
    if (i >= (long)Mrows * ld) return;
    int m = (int)(i / ld), n = (int)(i % ld);
    int r0 = fnode[fmess[m * 4 + 0]];
    int r1 = 64 + fmess[m * 4 + 2];
    int r2 = 68 + fmess[m * 4 + 3];
    out[i] = Wg[(size_t)r0 * ld + n] + Wg[(size_t)r1 * ld + n] + Wg[(size_t)r2 * ld + n] + bg[n];
}
// Fused LSTM gates: z[M,768] (pre-act), flog[M,3,256] (forget pre-act per nbr)
__global__ void k_lstm_pw(const float* __restrict__ z, const float* __restrict__ flog,
                          const float* __restrict__ c_old, const int* __restrict__ bgraph,
                          float* __restrict__ c_new, __bf16* __restrict__ h_new, int Mrows) {
    long i = (long)blockIdx.x * blockDim.x + threadIdx.x;
    if (i >= (long)Mrows * HDIM) return;
    int m = (int)(i >> 8), n = (int)(i & 255);
    float iv = sig_(z[(size_t)m * H3DIM + n]);
    float ov = sig_(z[(size_t)m * H3DIM + HDIM + n]);
    float uv = tanhf(z[(size_t)m * H3DIM + 2 * HDIM + n]);
    float acc = iv * uv;
#pragma unroll
    for (int d = 0; d < 3; ++d) {
        int j = bgraph[m * 3 + d];
        float fv = sig_(flog[((size_t)m * 3 + d) * HDIM + n]);
        acc += fv * c_old[(size_t)j * HDIM + n];
    }
    float h2 = ov * tanhf(acc);
    if (m == 0) { acc = 0.0f; h2 = 0.0f; }
    c_new[i] = acc;
    h_new[i] = (__bf16)h2;
}
// dst[m, off:off+256] = src[m, 0:256]  (bf16 concat pack)
__global__ void k_pack(const __bf16* __restrict__ src, __bf16* __restrict__ dst,
                       int Kd, int off, int rows) {
    long i = (long)blockIdx.x * blockDim.x + threadIdx.x;
    if (i >= (long)rows * HDIM) return;
    int m = (int)(i >> 8), n = (int)(i & 255);
    dst[(size_t)m * Kd + off + n] = src[i];
}

// ---------------------------------------------------------------------------
extern "C" void kernel_launch(void* const* d_in, const int* in_sizes, int n_in,
                              void* d_out, int out_size, void* d_ws, size_t ws_size,
                              hipStream_t stream) {
    (void)in_sizes; (void)n_in; (void)out_size; (void)ws_size;

    const int* fnode_tree   = (const int*)d_in[0];
    const int* fmess_tree   = (const int*)d_in[1];
    const int* agraph_tree  = (const int*)d_in[2];
    const int* bgraph_tree  = (const int*)d_in[3];
    const int* cgraph       = (const int*)d_in[4];
    const int* fnode_graph  = (const int*)d_in[5];
    const int* fmess_graph  = (const int*)d_in[6];
    const int* agraph_graph = (const int*)d_in[7];
    const int* bgraph_graph = (const int*)d_in[8];
    const float* Ec   = (const float*)d_in[9];
    const float* Ei   = (const float*)d_in[10];
    const float* El   = (const float*)d_in[11];
    const float* Wc_w = (const float*)d_in[12];
    const float* Wc_b = (const float*)d_in[13];
    const float* Wi_w = (const float*)d_in[14];
    const float* Wi_b = (const float*)d_in[15];

    struct Enc { const float *Wg, *bg, *Wf, *bf, *Wo, *bo; };
    Enc eg = {(const float*)d_in[16], (const float*)d_in[17], (const float*)d_in[18],
              (const float*)d_in[19], (const float*)d_in[20], (const float*)d_in[21]};
    Enc eb = {(const float*)d_in[22], (const float*)d_in[23], (const float*)d_in[24],
              (const float*)d_in[25], (const float*)d_in[26], (const float*)d_in[27]};
    Enc ef = {(const float*)d_in[28], (const float*)d_in[29], (const float*)d_in[30],
              (const float*)d_in[31], (const float*)d_in[32], (const float*)d_in[33]};
    Enc ei = {(const float*)d_in[34], (const float*)d_in[35], (const float*)d_in[36],
              (const float*)d_in[37], (const float*)d_in[38], (const float*)d_in[39]};

    float* out = (float*)d_out;  // [hnode NT*H | hinter NT*H | hbond NT*H | hatom NA*H]

    // ---- workspace carving ----
    char* wsB = (char*)d_ws;
    size_t offB = 0;
    auto alloc = [&](size_t bytes) -> void* {
        void* p = wsB + offB;
        offB += (bytes + 255) & ~(size_t)255;
        return p;
    };
    float*  zpage = (float*)alloc(256);                   // zero page for OOB async loads
    float*  zpre  = (float*)alloc((size_t)MA_ * H3DIM * 4);
    float*  zbuf  = (float*)alloc((size_t)MA_ * H3DIM * 4);
    float*  fpre  = (float*)alloc((size_t)MA_ * HDIM * 4);
    float*  flog  = (float*)alloc((size_t)MA_ * 3 * HDIM * 4);
    __bf16* sbuf  = (__bf16*)alloc((size_t)MA_ * HDIM * 2);
    __bf16* hmsg0 = (__bf16*)alloc((size_t)MA_ * HDIM * 2);
    __bf16* hmsg1 = (__bf16*)alloc((size_t)MA_ * HDIM * 2);
    float*  c0    = (float*)alloc((size_t)MA_ * HDIM * 4);
    float*  c1    = (float*)alloc((size_t)MA_ * HDIM * 4);
    __bf16* nei   = (__bf16*)alloc((size_t)NA_ * HDIM * 2);
    __bf16* rowtmp = (__bf16*)alloc((size_t)NT_ * HDIM * 2);
    __bf16* hnbuf  = (__bf16*)alloc((size_t)NT_ * HDIM * 2);
    __bf16* packA  = (__bf16*)alloc((size_t)NT_ * H3DIM * 2);
    __bf16* hnode_cur = (__bf16*)alloc((size_t)NT_ * HDIM * 2);
    __bf16* hatomB  = (__bf16*)alloc((size_t)NA_ * HDIM * 2);
    __bf16* hbondB  = (__bf16*)alloc((size_t)NT_ * HDIM * 2);
    __bf16* hinterB = (__bf16*)alloc((size_t)NT_ * HDIM * 2);
    // pre-transposed bf16 weights [N,K]
    __bf16* wgT_top = (__bf16*)alloc((size_t)H3DIM * HDIM * 2);
    __bf16* wgT_h   = (__bf16*)alloc((size_t)H3DIM * HDIM * 2);
    __bf16* wfT_top = (__bf16*)alloc((size_t)HDIM * HDIM * 2);
    __bf16* wfT_h   = (__bf16*)alloc((size_t)HDIM * HDIM * 2);
    __bf16* wT_big  = (__bf16*)alloc((size_t)HDIM * 3 * HDIM * 2);

    // ---- launch helpers ----
    const dim3 B256(256);
    auto ew = [](long n) { return dim3((unsigned)((n + 255) / 256)); };
    auto zf = [&](float* p, long n)  { k_zero_f<<<ew(n), B256, 0, stream>>>(p, n); };
    auto zh = [&](__bf16* p, long n) { k_zero_h<<<ew(n), B256, 0, stream>>>(p, n); };
    auto cvtT = [&](const float* s, __bf16* d, int K, int N) {
        k_cvt_t<<<ew((long)K * N), B256, 0, stream>>>(s, d, K, N);
    };
    auto gemm = [&](const __bf16* A, const __bf16* Agat, const int* gidx, int gs, int go,
                    const __bf16* Wt, const float* addm, int ldad,
                    const float* rowW, int ldrw, const int* ridx, int rs, int ro, int rb,
                    const float* bias, float* Cf, __bf16* Cb, int ldc,
                    int M, int N, int K, int relu, int mask0) {
        dim3 grid((M + BM - 1) / BM, N / BN);
        k_gemm<<<grid, B256, 0, stream>>>(A, Agat, gidx, gs, go, Wt, addm, ldad,
                                          rowW, ldrw, ridx, rs, ro, rb,
                                          bias, (const __bf16*)zpage, Cf, Cb, ldc,
                                          M, N, K, relu, mask0);
    };

    zf(zpage, 64);  // zero page must be initialized before any GEMM

    // LSTM depth loop (zpre/fpre + WgH/WfH already prepared); returns final h (bf16)
    auto run_lstm = [&](int M, const int* bgraph, int depth) -> __bf16* {
        zh(hmsg0, (long)M * HDIM);
        zf(c0,    (long)M * HDIM);
        __bf16 *hc = hmsg0, *hx = hmsg1;
        float  *cc = c0,    *cx = c1;
        for (int t = 0; t < depth; ++t) {
            k_gather_sum<<<ew((long)M * HDIM), B256, 0, stream>>>(hc, bgraph, 3, sbuf, M);
            // z = zpre + s @ WgH
            gemm(sbuf, nullptr, nullptr, 0, 0, wgT_h, zpre, H3DIM,
                 nullptr, 0, nullptr, 0, 0, 0, nullptr, zbuf, nullptr, H3DIM,
                 M, H3DIM, HDIM, 0, 0);
            // f_d = fpre + h[bgraph[:,d]] @ WfH
            for (int d = 0; d < 3; ++d)
                gemm(nullptr, hc, bgraph, 3, d, wfT_h, fpre, HDIM,
                     nullptr, 0, nullptr, 0, 0, 0, nullptr, flog + d * HDIM, nullptr, 3 * HDIM,
                     M, HDIM, HDIM, 0, 0);
            k_lstm_pw<<<ew((long)M * HDIM), B256, 0, stream>>>(zbuf, flog, cc, bgraph, cx, hx, M);
            __bf16* th = hc; hc = hx; hx = th;
            float*  tc = cc; cc = cx; cx = tc;
        }
        return hc;
    };

    // ======================= graph (atom) encoder =======================
    cvtT(eg.Wg + (size_t)88 * H3DIM, wgT_h, HDIM, H3DIM);  // recurrent rows 88..343 -> [768,256]
    cvtT(eg.Wf + (size_t)88 * HDIM,  wfT_h, HDIM, HDIM);
    cvtT(eg.Wo + (size_t)64 * HDIM,  wT_big, HDIM, HDIM);  // nei part of Wo -> [256,256]
    k_zpre_onehot<<<ew((long)MA_ * H3DIM), B256, 0, stream>>>(eg.Wg, eg.bg, fnode_graph,
                                                              fmess_graph, H3DIM, zpre, MA_);
    k_zpre_onehot<<<ew((long)MA_ * HDIM), B256, 0, stream>>>(eg.Wf, eg.bf, fnode_graph,
                                                             fmess_graph, HDIM, fpre, MA_);
    __bf16* hg = run_lstm(MA_, bgraph_graph, 3);
    k_gather_sum<<<ew((long)NA_ * HDIM), B256, 0, stream>>>(hg, agraph_graph, 4, nei, NA_);
    // hatom = relu( Wo[fnode] + nei @ WoH + bo ), row0 masked
    gemm(nei, nullptr, nullptr, 0, 0, wT_big, nullptr, 0,
         eg.Wo, HDIM, fnode_graph, 1, 0, 0, eg.bo,
         out + 3L * NT_ * HDIM, hatomB, HDIM, NA_, HDIM, HDIM, 1, 1);

    // ==================== tree encoder (shared logic) ====================
    auto tree_enc = [&](const Enc& e, const __bf16* hnode, float* out_f, __bf16* out_b) {
        cvtT(e.Wg, wgT_top, HDIM, H3DIM);
        cvtT(e.Wg + (size_t)276 * H3DIM, wgT_h, HDIM, H3DIM);
        cvtT(e.Wf, wfT_top, HDIM, HDIM);
        cvtT(e.Wf + (size_t)276 * HDIM,  wfT_h, HDIM, HDIM);
        cvtT(e.Wo, wT_big, 2 * HDIM, HDIM);
        // zpre = hnode[src] @ WgTop + Wg[256+pos] + bg ; fpre likewise
        gemm(nullptr, hnode, fmess_tree, 3, 0, wgT_top, nullptr, 0,
             e.Wg, H3DIM, fmess_tree, 3, 2, 256, e.bg, zpre, nullptr, H3DIM,
             MT_, H3DIM, HDIM, 0, 0);
        gemm(nullptr, hnode, fmess_tree, 3, 0, wfT_top, nullptr, 0,
             e.Wf, HDIM, fmess_tree, 3, 2, 256, e.bf, fpre, nullptr, HDIM,
             MT_, HDIM, HDIM, 0, 0);
        __bf16* h = run_lstm(MT_, bgraph_tree, 3);
        k_gather_sum<<<ew((long)NT_ * HDIM), B256, 0, stream>>>(h, agraph_tree, 4, nei, NT_);
        k_pack<<<ew((long)NT_ * HDIM), B256, 0, stream>>>(hnode, packA, 2 * HDIM, 0, NT_);
        k_pack<<<ew((long)NT_ * HDIM), B256, 0, stream>>>(nei,   packA, 2 * HDIM, HDIM, NT_);
        gemm(packA, nullptr, nullptr, 0, 0, wT_big, nullptr, 0,
             nullptr, 0, nullptr, 0, 0, 0, e.bo, out_f, out_b, HDIM,
             NT_, HDIM, 2 * HDIM, 1, 1);
    };

    // ---- encoder 'b' (bond/tree over El embedding) ----
    k_gather_rows<<<ew((long)NT_ * HDIM), B256, 0, stream>>>(El, fnode_tree, 3, 2, hnode_cur, NT_);
    tree_enc(eb, hnode_cur, out + 2L * NT_ * HDIM, hbondB);

    // ---- encoder 'f' (inter): hnode_f = relu([Ei[fnode], sum hatom[cgraph]] @ Wi + bi) ----
    k_gather_sum<<<ew((long)NT_ * HDIM), B256, 0, stream>>>(hatomB, cgraph, 8, hnbuf, NT_);
    k_gather_rows<<<ew((long)NT_ * HDIM), B256, 0, stream>>>(Ei, fnode_tree, 3, 1, rowtmp, NT_);
    k_pack<<<ew((long)NT_ * HDIM), B256, 0, stream>>>(rowtmp, packA, 2 * HDIM, 0, NT_);
    k_pack<<<ew((long)NT_ * HDIM), B256, 0, stream>>>(hnbuf,  packA, 2 * HDIM, HDIM, NT_);
    cvtT(Wi_w, wT_big, 2 * HDIM, HDIM);
    gemm(packA, nullptr, nullptr, 0, 0, wT_big, nullptr, 0,
         nullptr, 0, nullptr, 0, 0, 0, Wi_b, nullptr, hnode_cur, HDIM,
         NT_, HDIM, 2 * HDIM, 1, 0);
    tree_enc(ef, hnode_cur, out + 1L * NT_ * HDIM, hinterB);

    // ---- encoder 'i' (final): hnode_i = relu([Ec[fnode], hinter, hbond] @ Wc + bc) ----
    k_gather_rows<<<ew((long)NT_ * HDIM), B256, 0, stream>>>(Ec, fnode_tree, 3, 0, rowtmp, NT_);
    k_pack<<<ew((long)NT_ * HDIM), B256, 0, stream>>>(rowtmp,  packA, 3 * HDIM, 0, NT_);
    k_pack<<<ew((long)NT_ * HDIM), B256, 0, stream>>>(hinterB, packA, 3 * HDIM, HDIM, NT_);
    k_pack<<<ew((long)NT_ * HDIM), B256, 0, stream>>>(hbondB,  packA, 3 * HDIM, 2 * HDIM, NT_);
    cvtT(Wc_w, wT_big, 3 * HDIM, HDIM);
    gemm(packA, nullptr, nullptr, 0, 0, wT_big, nullptr, 0,
         nullptr, 0, nullptr, 0, 0, 0, Wc_b, nullptr, hnode_cur, HDIM,
         NT_, HDIM, 3 * HDIM, 1, 0);
    tree_enc(ei, hnode_cur, out, nullptr);
}
